// SamadhiCore_74002286510698
// MI455X (gfx1250) — matrix operand order repaired and checked by reading, not compiled, not run on hardware
//
#include <hip/hip_runtime.h>

#define DIM       512
#define DIMP      520     // padded xA row (halves): bank stride 4 -> conflict-free frags
#define HALF_DIM  256
#define NPROBES   1024
#define BATCH     65536
#define MTILE     32      // batch rows per workgroup
#define KQ        64      // staged K chunk (halves)
#define KQP       72      // padded Bs row (halves): bank stride 36 -> conflict-free frags
#define NPANEL    64      // staged N panel
#define NWAVES    8
#define TEMP      0.2f
#define GATE_TH   0.1f
#define LN_EPS    1e-5f

typedef __attribute__((ext_vector_type(16))) __bf16 v16bf;
typedef __attribute__((ext_vector_type(8)))  __bf16 v8bf;
typedef __attribute__((ext_vector_type(8)))  float  v8f;

__device__ __forceinline__ v16bf cat8(v8bf lo, v8bf hi) {
    return __builtin_shufflevector(lo, hi, 0,1,2,3,4,5,6,7,8,9,10,11,12,13,14,15);
}

// LDS byte offset of a __shared__ object (generic -> addrspace(3) -> int)
__device__ __forceinline__ unsigned lds_offset_of(const void* p) {
    return (unsigned)(unsigned long long)((__attribute__((address_space(3))) const char*)p);
}

__device__ __forceinline__ void wait_async_lds() {
    asm volatile("s_wait_asynccnt 0x0" ::: "memory");
}

// Stage one 64-row x 64-half B panel chunk via the async DMA path:
// global_load_async_to_lds_b128 — no VGPR round trip, tracked by ASYNCcnt.
// 2 x 16B per thread (256 threads). Source is bf16 row-major [N][ldK].
__device__ __forceinline__ void stage_panel_async(__bf16 (* __restrict__ Bsb)[KQP],
        const __bf16* __restrict__ src, int ldK, int n0, int k0, int tid)
{
    #pragma unroll
    for (int it = 0; it < 2; ++it) {
        int idx = it * 256 + tid;
        int kc  = idx & 7;            // 8 chunks of 8 halves per row
        int j   = idx >> 3;           // 64 rows
        const __bf16* gp = src + (size_t)(n0 + j) * ldK + k0 + kc * 8;
        unsigned lds_off = lds_offset_of(&Bsb[j][kc * 8]);
        asm volatile("global_load_async_to_lds_b128 %0, %1, off"
                     :: "v"(lds_off), "v"(gp) : "memory");
    }
}

// Per-wave: one 16x16 tile, accumulate a 64-deep K chunk (2 x wmma 16x16x32 bf16).
__device__ __forceinline__ v8f wmma_k64(v8f acc,
        const __bf16* __restrict__ aRow, const __bf16* __restrict__ bRow, int lane)
{
    const int r = lane & 15;
    const int h = lane >> 4;
    #pragma unroll
    for (int kk = 0; kk < 2; ++kk) {
        const int k = kk * 32;
        v8bf a0 = *(const v8bf*)(aRow + (size_t)r * DIMP + k + h * 8);
        v8bf a1 = *(const v8bf*)(aRow + (size_t)r * DIMP + k + 16 + h * 8);
        v8bf b0 = *(const v8bf*)(bRow + (size_t)r * KQP + k + h * 8);
        v8bf b1 = *(const v8bf*)(bRow + (size_t)r * KQP + k + 16 + h * 8);
        acc = __builtin_amdgcn_wmma_f32_16x16x32_bf16(
                false, cat8(a0, a1), false, cat8(b0, b1), (short)0, acc, false, false);
    }
    return acc;
}

// Full K reduction for one 64-col panel with double-buffered async staging.
// Async copy for chunk k+1 is issued, WMMAs consume chunk k, then the wave
// waits ASYNCcnt==0 only at the barrier. Uniform control flow (barriers inside).
__device__ __forceinline__ v8f gemm_panel(__bf16 (* __restrict__ Bs2)[NPANEL][KQP],
        const __bf16* __restrict__ Wt, int ldK, int nkq, int n0,
        const __bf16* __restrict__ aRow, int ns, int tid, int lane)
{
    v8f acc = {};
    stage_panel_async(Bs2[0], Wt, ldK, n0, 0, tid);
    wait_async_lds();
    __syncthreads();
    int buf = 0;
    for (int kq = 0; kq < nkq; ++kq) {
        if (kq + 1 < nkq) stage_panel_async(Bs2[buf ^ 1], Wt, ldK, n0, (kq + 1) * KQ, tid);
        acc = wmma_k64(acc, aRow + kq * KQ, &Bs2[buf][ns * 16][0], lane);
        wait_async_lds();
        __syncthreads();
        buf ^= 1;
    }
    return acc;
}

// One-time: convert weights to bf16 (transposed to N-major where needed) in ws.
__global__ void __launch_bounds__(256)
samadhi_prep_kernel(const float* __restrict__ Wa, const float* __restrict__ probes,
                    const float* __restrict__ W1, const float* __restrict__ W2,
                    __bf16* __restrict__ WaT, __bf16* __restrict__ probesB,
                    __bf16* __restrict__ W1T, __bf16* __restrict__ W2T)
{
    int tid = blockIdx.x * 256 + threadIdx.x;
    int stride = gridDim.x * 256;
    for (int i = tid; i < DIM * DIM; i += stride) {           // WaT[n][k] = Wa[k][n]
        int n = i / DIM, k = i - n * DIM;
        WaT[i] = (__bf16)Wa[(size_t)k * DIM + n];
    }
    for (int i = tid; i < NPROBES * DIM; i += stride)         // probes already [n][k]
        probesB[i] = (__bf16)probes[i];
    for (int i = tid; i < HALF_DIM * DIM; i += stride) {      // W1T[n][k], n<256,k<512
        int n = i / DIM, k = i - n * DIM;
        W1T[i] = (__bf16)W1[(size_t)k * HALF_DIM + n];
    }
    for (int i = tid; i < DIM * HALF_DIM; i += stride) {      // W2T[n][k], n<512,k<256
        int n = i / HALF_DIM, k = i - n * HALF_DIM;
        W2T[i] = (__bf16)W2[(size_t)k * DIM + n];
    }
}

__global__ void __launch_bounds__(256)
samadhi_fused_kernel(const float* __restrict__ x,
                     const float* __restrict__ probes,      // f32, for winner gather
                     const __bf16* __restrict__ WaT,
                     const __bf16* __restrict__ probesB,
                     const __bf16* __restrict__ W1T,
                     const __bf16* __restrict__ W2T,
                     const float* __restrict__ ba,  const float* __restrict__ ga,
                     const float* __restrict__ bta, const float* __restrict__ b1,
                     const float* __restrict__ g1,  const float* __restrict__ bt1,
                     const float* __restrict__ b2,
                     float* __restrict__ out_s,     float* __restrict__ out_probs,
                     int* __restrict__ out_winner,  float* __restrict__ out_conf,
                     float* __restrict__ ws_zmax,   float* __restrict__ ws_l)
{
    __shared__ __align__(16) __bf16 xA[MTILE][DIMP];          // 32.5 KB  activation (A operand)
    __shared__ __align__(16) __bf16 Bs[2][NPANEL][KQP];       // 18 KB    double-buffered B panel
    __shared__ __align__(16) float  yF[MTILE][DIM];           // 64 KB    f32 state
    __shared__ __align__(16) float  hF[MTILE][HALF_DIM];      // 32 KB    f32 scratch
    __shared__ float m_run[MTILE], l_run[MTILE], gatev[MTILE];
    __shared__ int   winv[MTILE];

    const int tid  = threadIdx.x;
    const int lane = tid & 31;
    const int wv   = tid >> 5;
    const int mt   = wv & 1;        // 2 m-tiles of 16 rows
    const int ns   = wv >> 1;       // 4 n-subtiles within a 64-col panel
    const int row0 = blockIdx.x * MTILE;
    const __bf16* aRow = &xA[mt * 16][0];

    // ---------------- Phase A: load x tile, convert to bf16 A operand ----
    for (int it = 0; it < (MTILE * DIM) / (256 * 4); ++it) {  // 16 iters, 4 floats each
        int idx = (it * 256 + tid) * 4;
        int r = idx >> 9, c = idx & 511;
        float4 f = *(const float4*)(x + (size_t)(row0 + r) * DIM + c);
        xA[r][c + 0] = (__bf16)f.x; xA[r][c + 1] = (__bf16)f.y;
        xA[r][c + 2] = (__bf16)f.z; xA[r][c + 3] = (__bf16)f.w;
    }
    __syncthreads();

    // ---------------- Phase B: y = x @ Wa ---------------------------------
    for (int np = 0; np < DIM / NPANEL; ++np) {
        v8f acc = gemm_panel(Bs, WaT, DIM, DIM / KQ, np * NPANEL, aRow, ns, tid, lane);
        int c  = np * NPANEL + ns * 16 + (lane & 15);
        int rb = mt * 16 + (lane >> 4) * 8;
        #pragma unroll
        for (int g = 0; g < 8; ++g) yF[rb + g][c] = acc[g];
    }
    __syncthreads();

    // ---------------- Phase C: LayerNorm + tanh + L2-normalize ------------
    for (int r = wv; r < MTILE; r += NWAVES) {
        float s1 = 0.f, s2 = 0.f, v[16];
        #pragma unroll
        for (int i = 0; i < 16; ++i) {
            int c = lane + i * 32;
            float t = yF[r][c] + ba[c];
            v[i] = t; s1 += t; s2 += t * t;
        }
        #pragma unroll
        for (int m = 16; m >= 1; m >>= 1) { s1 += __shfl_xor(s1, m, 32); s2 += __shfl_xor(s2, m, 32); }
        float mu  = s1 * (1.0f / DIM);
        float var = s2 * (1.0f / DIM) - mu * mu;
        float inv = rsqrtf(var + LN_EPS);
        float nrm = 0.f;
        #pragma unroll
        for (int i = 0; i < 16; ++i) {
            int c = lane + i * 32;
            float t = tanhf((v[i] - mu) * inv * ga[c] + bta[c]);
            v[i] = t; nrm += t * t;
        }
        #pragma unroll
        for (int m = 16; m >= 1; m >>= 1) nrm += __shfl_xor(nrm, m, 32);
        float rinv = rsqrtf(nrm);
        #pragma unroll
        for (int i = 0; i < 16; ++i) {
            int c = lane + i * 32;
            yF[r][c] = v[i];                    // x_adapted (f32)
            xA[r][c] = (__bf16)(v[i] * rinv);   // x_norm (bf16 A operand)
        }
    }
    if (tid < MTILE) { m_run[tid] = -1e30f; l_run[tid] = 0.f; winv[tid] = 0; }
    __syncthreads();

    // ---------------- Phase D: scores + online softmax --------------------
    for (int np = 0; np < NPROBES / NPANEL; ++np) {
        v8f acc = gemm_panel(Bs, probesB, DIM, DIM / KQ, np * NPANEL, aRow, ns, tid, lane);
        int cl = ns * 16 + (lane & 15);
        int rb = mt * 16 + (lane >> 4) * 8;
        #pragma unroll
        for (int g = 0; g < 8; ++g) hF[rb + g][cl] = acc[g];
        __syncthreads();
        // stream raw scores to global (normalized by second kernel)
        for (int it = 0; it < (MTILE * NPANEL) / (256 * 4); ++it) {  // 2 iters, float4
            int idx = (it * 256 + tid) * 4;
            int r = idx >> 6, c = idx & 63;
            float4 f; f.x = hF[r][c]; f.y = hF[r][c + 1]; f.z = hF[r][c + 2]; f.w = hF[r][c + 3];
            *(float4*)(out_probs + (size_t)(row0 + r) * NPROBES + np * NPANEL + c) = f;
        }
        // running max / sumexp / argmax per row (wave per row)
        for (int r = wv; r < MTILE; r += NWAVES) {
            float z0 = hF[r][lane]      * (1.0f / TEMP);
            float z1 = hF[r][lane + 32] * (1.0f / TEMP);
            float zm = fmaxf(z0, z1);
            int   zi = (z1 > z0) ? (lane + 32) : lane;
            #pragma unroll
            for (int m = 16; m >= 1; m >>= 1) {
                float om = __shfl_xor(zm, m, 32);
                int   oi = __shfl_xor(zi, m, 32);
                if (om > zm) { zm = om; zi = oi; }
            }
            float oldm = m_run[r];
            float newm = fmaxf(oldm, zm);
            float contrib = __expf(z0 - newm) + __expf(z1 - newm);
            #pragma unroll
            for (int m = 16; m >= 1; m >>= 1) contrib += __shfl_xor(contrib, m, 32);
            if (lane == 0) {
                l_run[r] = l_run[r] * __expf(oldm - newm) + contrib;
                m_run[r] = newm;
                if (zm > oldm) winv[r] = np * NPANEL + zi;
            }
        }
        __syncthreads();
    }

    // ---------------- finalize winner / confidence / gate, build s0 -------
    for (int r = tid; r < MTILE; r += 256) {
        float zm = m_run[r], l = l_run[r];
        ws_zmax[row0 + r]    = zm;
        ws_l[row0 + r]       = l;
        out_winner[row0 + r] = winv[r];
        out_conf[row0 + r]   = 1.0f / l;
        gatev[r] = ((zm * TEMP) > GATE_TH) ? 1.0f : 0.0f;
    }
    __syncthreads();
    for (int it = 0; it < (MTILE * DIM) / 256; ++it) {
        int idx = it * 256 + tid;
        int r = idx >> 9, c = idx & 511;
        float s0v = gatev[r] * (0.5f * yF[r][c] + 0.5f * probes[(size_t)winv[r] * DIM + c]);
        yF[r][c] = s0v;
        xA[r][c] = (__bf16)s0v;
    }
    __syncthreads();

    // ---------------- Phase E: 6 refinement steps --------------------------
    for (int step = 0; step < 6; ++step) {
        // h_raw = s @ W1
        for (int np = 0; np < HALF_DIM / NPANEL; ++np) {
            v8f acc = gemm_panel(Bs, W1T, DIM, DIM / KQ, np * NPANEL, aRow, ns, tid, lane);
            int c  = np * NPANEL + ns * 16 + (lane & 15);
            int rb = mt * 16 + (lane >> 4) * 8;
            #pragma unroll
            for (int g = 0; g < 8; ++g) hF[rb + g][c] = acc[g];
        }
        __syncthreads();
        // h = relu(LN(h_raw)) -> bf16 A operand (cols 0..255)
        for (int r = wv; r < MTILE; r += NWAVES) {
            float s1 = 0.f, s2 = 0.f, v[8];
            #pragma unroll
            for (int i = 0; i < 8; ++i) {
                int c = lane + i * 32;
                float t = hF[r][c] + b1[c];
                v[i] = t; s1 += t; s2 += t * t;
            }
            #pragma unroll
            for (int m = 16; m >= 1; m >>= 1) { s1 += __shfl_xor(s1, m, 32); s2 += __shfl_xor(s2, m, 32); }
            float mu  = s1 * (1.0f / HALF_DIM);
            float var = s2 * (1.0f / HALF_DIM) - mu * mu;
            float inv = rsqrtf(var + LN_EPS);
            #pragma unroll
            for (int i = 0; i < 8; ++i) {
                int c = lane + i * 32;
                float t = (v[i] - mu) * inv * g1[c] + bt1[c];
                xA[r][c] = (__bf16)fmaxf(t, 0.0f);
            }
        }
        __syncthreads();
        // s = 0.7*s + 0.3*tanh(h @ W2 + b2)   (K=256)
        for (int np = 0; np < DIM / NPANEL; ++np) {
            v8f acc = gemm_panel(Bs, W2T, HALF_DIM, HALF_DIM / KQ, np * NPANEL, aRow, ns, tid, lane);
            int c  = np * NPANEL + ns * 16 + (lane & 15);
            int rb = mt * 16 + (lane >> 4) * 8;
            #pragma unroll
            for (int g = 0; g < 8; ++g) {
                float sv = yF[rb + g][c];
                yF[rb + g][c] = 0.7f * sv + 0.3f * tanhf(acc[g] + b2[c]);
            }
        }
        __syncthreads();
        // refresh bf16 A operand with new s
        for (int it = 0; it < (MTILE * DIM) / 256; ++it) {
            int idx = it * 256 + tid;
            int r = idx >> 9, c = idx & 511;
            xA[r][c] = (__bf16)yF[r][c];
        }
        __syncthreads();
    }

    // ---------------- write s_final ----------------------------------------
    for (int it = 0; it < (MTILE * DIM) / (256 * 4); ++it) {
        int idx = (it * 256 + tid) * 4;
        int r = idx >> 9, c = idx & 511;
        float4 f; f.x = yF[r][c]; f.y = yF[r][c + 1]; f.z = yF[r][c + 2]; f.w = yF[r][c + 3];
        *(float4*)(out_s + (size_t)(row0 + r) * DIM + c) = f;
    }
}

// Normalize raw scores -> softmax probs using per-row stats from the fused kernel.
__global__ void __launch_bounds__(256)
samadhi_softmax_kernel(float* __restrict__ probs,
                       const float* __restrict__ zmax,
                       const float* __restrict__ lsum)
{
    int row = blockIdx.x;
    float zm  = zmax[row];
    float inv = 1.0f / lsum[row];
    size_t base = (size_t)row * NPROBES;
    for (int i = threadIdx.x; i < NPROBES; i += 256) {
        float raw = probs[base + i];
        probs[base + i] = __expf(raw * (1.0f / TEMP) - zm) * inv;
    }
}

extern "C" void kernel_launch(void* const* d_in, const int* in_sizes, int n_in,
                              void* d_out, int out_size, void* d_ws, size_t ws_size,
                              hipStream_t stream) {
    const float* x      = (const float*)d_in[0];
    const float* probes = (const float*)d_in[1];
    const float* Wa     = (const float*)d_in[2];
    const float* ba     = (const float*)d_in[3];
    const float* ga     = (const float*)d_in[4];
    const float* bta    = (const float*)d_in[5];
    const float* W1     = (const float*)d_in[6];
    const float* b1     = (const float*)d_in[7];
    const float* g1     = (const float*)d_in[8];
    const float* bt1    = (const float*)d_in[9];
    const float* W2     = (const float*)d_in[10];
    const float* b2     = (const float*)d_in[11];

    float* out       = (float*)d_out;
    float* out_s     = out;                                           // [B,512]
    float* out_probs = out + (size_t)BATCH * DIM;                     // [B,1024]
    int*   out_win   = (int*)(out + (size_t)BATCH * (DIM + NPROBES)); // [B] int32
    float* out_conf  = out + (size_t)BATCH * (DIM + NPROBES) + BATCH; // [B]

    // ws layout: f32 stats, then bf16 weight copies
    float*  ws_zmax  = (float*)d_ws;                                   // [B]
    float*  ws_l     = ws_zmax + BATCH;                                // [B]
    __bf16* WaT      = (__bf16*)(ws_l + BATCH);                        // [512][512]
    __bf16* probesB  = WaT + (size_t)DIM * DIM;                        // [1024][512]
    __bf16* W1T      = probesB + (size_t)NPROBES * DIM;                // [256][512]
    __bf16* W2T      = W1T + (size_t)HALF_DIM * DIM;                   // [512][256]

    dim3 block(256);
    samadhi_prep_kernel<<<dim3(256), block, 0, stream>>>(
        Wa, probes, W1, W2, WaT, probesB, W1T, W2T);

    samadhi_fused_kernel<<<dim3(BATCH / MTILE), block, 0, stream>>>(
        x, probes, WaT, probesB, W1T, W2T,
        ba, ga, bta, b1, g1, bt1, b2,
        out_s, out_probs, out_win, out_conf, ws_zmax, ws_l);

    samadhi_softmax_kernel<<<dim3(BATCH), block, 0, stream>>>(out_probs, ws_zmax, ws_l);
}
